// GIN_66915590472498
// MI455X (gfx1250) — matrix operand order repaired and checked by reading, not compiled, not run on hardware
//
#include <hip/hip_runtime.h>
#include <hip/hip_bf16.h>

#define D_FEAT 128
#define WPAD 136          // LDS row stride in bf16 elements (padded: 272B = 68 banks)
#define ROWS_PER_WG 64

typedef __attribute__((ext_vector_type(16))) __bf16 v16bf;
typedef __attribute__((ext_vector_type(8)))  float  v8f;

// ---------------- zero fill ----------------
__global__ void zero_kernel(float* __restrict__ p, size_t n) {
  size_t i = (size_t)blockIdx.x * blockDim.x + threadIdx.x;
  size_t stride = (size_t)gridDim.x * blockDim.x;
  for (; i < n; i += stride) p[i] = 0.0f;
}

// ---------------- edge scatter: agg[dst] += h[src]; one wave per edge ----------------
__global__ void scatter_kernel(const float* __restrict__ h,
                               const int* __restrict__ src,
                               const int* __restrict__ dst,
                               float* __restrict__ agg, int nedges) {
  size_t gtid = (size_t)blockIdx.x * blockDim.x + threadIdx.x;
  int wid = (int)(gtid >> 5);           // edge id
  int lane = threadIdx.x & 31;          // 4 floats per lane covers D=128
  if (wid >= nedges) return;
  long s = src[wid];
  long d = dst[wid];
  const float4 v = *(const float4*)(h + s * D_FEAT + lane * 4);
  float* p = agg + d * D_FEAT + lane * 4;
  atomicAdd(p + 0, v.x);
  atomicAdd(p + 1, v.y);
  atomicAdd(p + 2, v.z);
  atomicAdd(p + 3, v.w);
}

// Build WMMA A-operand (16-bit 16x32 layout) from z = agg + x, K chunk base kb.
// Lane half h: K segments [kb, kb+8) and [kb+16, kb+24) with kb already offset by 8*h.
__device__ inline v16bf pack_a_sum(const float* __restrict__ a,
                                   const float* __restrict__ x,
                                   long rowbase, int kb) {
  float4 a0 = *(const float4*)(a + rowbase + kb);
  float4 a1 = *(const float4*)(a + rowbase + kb + 4);
  float4 a2 = *(const float4*)(a + rowbase + kb + 16);
  float4 a3 = *(const float4*)(a + rowbase + kb + 20);
  float4 x0 = *(const float4*)(x + rowbase + kb);
  float4 x1 = *(const float4*)(x + rowbase + kb + 4);
  float4 x2 = *(const float4*)(x + rowbase + kb + 16);
  float4 x3 = *(const float4*)(x + rowbase + kb + 20);
  v16bf r;
  r[0]  = (__bf16)(a0.x + x0.x); r[1]  = (__bf16)(a0.y + x0.y);
  r[2]  = (__bf16)(a0.z + x0.z); r[3]  = (__bf16)(a0.w + x0.w);
  r[4]  = (__bf16)(a1.x + x1.x); r[5]  = (__bf16)(a1.y + x1.y);
  r[6]  = (__bf16)(a1.z + x1.z); r[7]  = (__bf16)(a1.w + x1.w);
  r[8]  = (__bf16)(a2.x + x2.x); r[9]  = (__bf16)(a2.y + x2.y);
  r[10] = (__bf16)(a2.z + x2.z); r[11] = (__bf16)(a2.w + x2.w);
  r[12] = (__bf16)(a3.x + x3.x); r[13] = (__bf16)(a3.y + x3.y);
  r[14] = (__bf16)(a3.z + x3.z); r[15] = (__bf16)(a3.w + x3.w);
  return r;
}

__device__ inline v16bf load_a_lds(const __bf16* __restrict__ u, int row, int kb) {
  v16bf r;
#pragma unroll
  for (int j = 0; j < 8; ++j) r[j] = u[row * WPAD + kb + j];
#pragma unroll
  for (int j = 0; j < 8; ++j) r[8 + j] = u[row * WPAD + kb + 16 + j];
  return r;
}

// B-operand: lane n holds K [kb, kb+8) and [kb+16, kb+24) from column-major LDS weights.
__device__ inline v16bf load_b_lds(const __bf16* __restrict__ wT, int n, int kb) {
  v16bf r;
#pragma unroll
  for (int j = 0; j < 16; ++j) r[j] = wT[n * WPAD + kb + (j < 8 ? j : j + 8)];
  return r;
}

// ---------------- fused GIN MLP: out = relu?(relu((agg+x)@wa+ba) @ wb + bb) ----------------
// 64 rows/WG, 8 waves: wave -> (mtile = w>>1, column half = w&1).
__global__ __launch_bounds__(256)
void gin_mlp_kernel(const float* __restrict__ agg,
                    const float* __restrict__ xin,
                    const float* __restrict__ wa, const float* __restrict__ ba,
                    const float* __restrict__ wb, const float* __restrict__ bb,
                    float* __restrict__ out, int nrows, int relu_out) {
  __shared__ __bf16 wT[D_FEAT * WPAD];       // weights, column-major [n][k], bf16
  __shared__ __bf16 u[ROWS_PER_WG * WPAD];   // relu(layer1) intermediate, row-major

  const int tid  = threadIdx.x;
  const int lane = tid & 31;
  const int wave = tid >> 5;
  const int row0 = blockIdx.x * ROWS_PER_WG;

  // stage layer-1 weights: global [k][n] f32 -> LDS [n][k] bf16
  for (int idx = tid; idx < D_FEAT * D_FEAT; idx += 256) {
    int k = idx >> 7, n = idx & 127;
    wT[n * WPAD + k] = (__bf16)wa[idx];
  }
  __syncthreads();

  const int mt    = wave >> 1;           // 0..3 -> 16-row tile
  const int nh    = wave & 1;            // 0..1 -> 64-column half
  const int khalf = lane >> 4;           // lane half selects K sub-range
  const int ncol  = lane & 15;           // C/D column for this lane
  const int mrow  = mt * 16 + (lane & 15);   // A row for this lane
  const int arow  = row0 + mrow;
  const int crow  = arow < nrows ? arow : nrows - 1;  // clamp tail reads
  const long rowbase = (long)crow * D_FEAT;

  v16bf A[4];
#pragma unroll
  for (int c = 0; c < 4; ++c)
    A[c] = pack_a_sum(agg, xin, rowbase, c * 32 + khalf * 8);

  // ---- layer 1: u = relu(z @ wa + ba) ----
#pragma unroll
  for (int nt = 0; nt < 4; ++nt) {
    const int n0 = nh * 64 + nt * 16;
    v8f acc = {};
#pragma unroll
    for (int c = 0; c < 4; ++c) {
      v16bf B = load_b_lds(wT, n0 + ncol, c * 32 + khalf * 8);
      acc = __builtin_amdgcn_wmma_f32_16x16x32_bf16(false, A[c], false, B,
                                                    (short)0, acc, false, false);
    }
    float bias = ba[n0 + ncol];
#pragma unroll
    for (int r = 0; r < 8; ++r) {
      float v = acc[r] + bias;
      v = v > 0.0f ? v : 0.0f;
      int m = mt * 16 + khalf * 8 + r;   // C/D row layout
      u[m * WPAD + n0 + ncol] = (__bf16)v;
    }
  }
  __syncthreads();

  // stage layer-2 weights into same LDS buffer
  for (int idx = tid; idx < D_FEAT * D_FEAT; idx += 256) {
    int k = idx >> 7, n = idx & 127;
    wT[n * WPAD + k] = (__bf16)wb[idx];
  }
  __syncthreads();

  v16bf A2[4];
#pragma unroll
  for (int c = 0; c < 4; ++c)
    A2[c] = load_a_lds(u, mrow, c * 32 + khalf * 8);

  // ---- layer 2: out = u @ wb + bb (+ optional relu) ----
  const int srow0 = row0 + mt * 16 + khalf * 8;
#pragma unroll
  for (int nt = 0; nt < 4; ++nt) {
    const int n0 = nh * 64 + nt * 16;
    v8f acc = {};
#pragma unroll
    for (int c = 0; c < 4; ++c) {
      v16bf B = load_b_lds(wT, n0 + ncol, c * 32 + khalf * 8);
      acc = __builtin_amdgcn_wmma_f32_16x16x32_bf16(false, A2[c], false, B,
                                                    (short)0, acc, false, false);
    }
    float bias = bb[n0 + ncol];
#pragma unroll
    for (int r = 0; r < 8; ++r) {
      int srow = srow0 + r;
      if (srow < nrows) {
        float v = acc[r] + bias;
        if (relu_out) v = v > 0.0f ? v : 0.0f;
        out[(long)srow * D_FEAT + n0 + ncol] = v;
      }
    }
  }
}

// ---------------- BatchNorm (training-mode) ----------------
__global__ void bn_stats_kernel(const float* __restrict__ h,
                                float* __restrict__ stats, int nrows) {
  int col = threadIdx.x;  // blockDim == 128
  float s = 0.0f, s2 = 0.0f;
  for (int r = blockIdx.x; r < nrows; r += gridDim.x) {
    float v = h[(long)r * D_FEAT + col];
    s += v; s2 += v * v;
  }
  atomicAdd(&stats[col], s);
  atomicAdd(&stats[D_FEAT + col], s2);
}

__global__ void bn_finalize_kernel(float* __restrict__ stats,
                                   const float* __restrict__ gamma,
                                   const float* __restrict__ beta, int nrows) {
  int col = threadIdx.x;  // 128 threads
  float inv_n = 1.0f / (float)nrows;
  float mean = stats[col] * inv_n;
  float var  = stats[D_FEAT + col] * inv_n - mean * mean;
  float scale = gamma[col] * rsqrtf(var + 1e-5f);
  stats[2 * D_FEAT + col] = scale;
  stats[3 * D_FEAT + col] = beta[col] - mean * scale;
}

__global__ void bn_apply_kernel(float* __restrict__ h,
                                const float* __restrict__ stats, size_t n) {
  size_t i = (size_t)blockIdx.x * blockDim.x + threadIdx.x;
  size_t stride = (size_t)gridDim.x * blockDim.x;
  for (; i < n; i += stride) {
    int col = (int)(i & (D_FEAT - 1));
    h[i] = h[i] * stats[2 * D_FEAT + col] + stats[3 * D_FEAT + col];
  }
}

extern "C" void kernel_launch(void* const* d_in, const int* in_sizes, int n_in,
                              void* d_out, int out_size, void* d_ws, size_t ws_size,
                              hipStream_t stream) {
  const float* x      = (const float*)d_in[0];
  const int*   edge   = (const int*)d_in[1];
  const float* w1a    = (const float*)d_in[2];
  const float* b1a    = (const float*)d_in[3];
  const float* w1b    = (const float*)d_in[4];
  const float* b1b    = (const float*)d_in[5];
  const float* gamma2 = (const float*)d_in[6];
  const float* beta2  = (const float*)d_in[7];
  const float* w2a    = (const float*)d_in[8];
  const float* b2a    = (const float*)d_in[9];
  const float* w2b    = (const float*)d_in[10];
  const float* b2b    = (const float*)d_in[11];

  const int nrows  = in_sizes[0] / D_FEAT;
  const int nedges = in_sizes[1] / 2;
  const int* src = edge;
  const int* dst = edge + nedges;

  // workspace: agg (N*D) | h (N*D) | stats (512 floats)
  float* agg   = (float*)d_ws;
  float* h     = agg + (size_t)nrows * D_FEAT;
  float* stats = h + (size_t)nrows * D_FEAT;
  float* out   = (float*)d_out;

  const size_t nd = (size_t)nrows * D_FEAT;
  const int sblocks = (int)(((size_t)nedges * 32 + 255) / 256);
  const int mblocks = (nrows + ROWS_PER_WG - 1) / ROWS_PER_WG;

  // --- conv 1 ---
  zero_kernel<<<2048, 256, 0, stream>>>(agg, nd);
  zero_kernel<<<1, 256, 0, stream>>>(stats, 4 * D_FEAT);
  scatter_kernel<<<sblocks, 256, 0, stream>>>(x, src, dst, agg, nedges);
  gin_mlp_kernel<<<mblocks, 256, 0, stream>>>(agg, x, w1a, b1a, w1b, b1b,
                                              h, nrows, 1);
  // --- batchnorm (training mode) ---
  bn_stats_kernel<<<1024, 128, 0, stream>>>(h, stats, nrows);
  bn_finalize_kernel<<<1, 128, 0, stream>>>(stats, gamma2, beta2, nrows);
  bn_apply_kernel<<<2048, 256, 0, stream>>>(h, stats, nd);

  // --- conv 2 ---
  zero_kernel<<<2048, 256, 0, stream>>>(agg, nd);
  scatter_kernel<<<sblocks, 256, 0, stream>>>(h, src, dst, agg, nedges);
  gin_mlp_kernel<<<mblocks, 256, 0, stream>>>(agg, h, w2a, b2a, w2b, b2b,
                                              out, nrows, 0);
}
